// MultiHeadAttention_27393301414383
// MI455X (gfx1250) — matrix operand order repaired and checked by reading, not compiled, not run on hardware
//
#include <hip/hip_runtime.h>
#include <hip/hip_bf16.h>

#define BDIM 4
#define SEQN 1024
#define DIMC 768
#define NH 12
#define HD 64
#define GPC 64
#define QSCALE 0.125f

typedef __bf16 bf16;
typedef __attribute__((ext_vector_type(16))) __bf16 v16bf;
typedef __attribute__((ext_vector_type(8)))  float  v8f;
typedef __attribute__((ext_vector_type(4)))  unsigned u32x4;
typedef __attribute__((ext_vector_type(8)))  unsigned u32x8;

// ---------------------------------------------------------------------------
// WMMA helpers (CDNA5 V_WMMA_F32_16X16X32_BF16, wave32)
// ---------------------------------------------------------------------------
__device__ __forceinline__ v8f wmma_bf16(v16bf a, v16bf b, v8f c) {
  return __builtin_amdgcn_wmma_f32_16x16x32_bf16(
      /*neg_a=*/false, a, /*neg_b=*/false, b,
      /*c_mod=*/(short)0, c, /*reuse_a=*/false, /*reuse_b=*/false);
}

// A fragment: 16(M) x 32(K) tile, row-major src[m*ld + k]
__device__ __forceinline__ v16bf load_a(const bf16* src, int ld) {
  int lane = threadIdx.x & 31;
  int m  = lane & 15;
  int hh = lane >> 4;
  const bf16* row = src + (size_t)m * ld;
  v16bf a;
#pragma unroll
  for (int j = 0; j < 8; ++j) {
    int k = 16 * hh + 2 * j;
    a[2 * j]     = row[k];
    a[2 * j + 1] = row[k + 1];
  }
  return a;
}

// B fragment: 32(K) x 16(N), K-major source: B[k][n] = src[k*ld + n]
__device__ __forceinline__ v16bf load_b(const bf16* src, int ld) {
  int lane = threadIdx.x & 31;
  int n  = lane & 15;
  int hh = lane >> 4;
  v16bf b;
#pragma unroll
  for (int j = 0; j < 8; ++j) {
    int k = 16 * hh + 2 * j;
    b[2 * j]     = src[(size_t)k * ld + n];
    b[2 * j + 1] = src[(size_t)(k + 1) * ld + n];
  }
  return b;
}

// B fragment from N-major storage: B[k][n] = src[n*ld + k]
__device__ __forceinline__ v16bf load_bt(const bf16* src, int ld) {
  int lane = threadIdx.x & 31;
  int n  = lane & 15;
  int hh = lane >> 4;
  const bf16* row = src + (size_t)n * ld;
  v16bf b;
#pragma unroll
  for (int j = 0; j < 8; ++j) {
    int k = 16 * hh + 2 * j;
    b[2 * j]     = row[k];
    b[2 * j + 1] = row[k + 1];
  }
  return b;
}

__device__ __forceinline__ void store_d_f32(float* dst, int ld, v8f d) {
  int lane = threadIdx.x & 31;
  int n  = lane & 15;
  int hh = lane >> 4;
#pragma unroll
  for (int r = 0; r < 8; ++r) dst[(size_t)(r + 8 * hh) * ld + n] = d[r];
}

// ---------------------------------------------------------------------------
// Tensor Data Mover: 2-D bf16 tile, global -> LDS (D# per ISA cdna5 §8.3/8.4)
// Issued once per wave; EXEC ignored by hardware. 2-group form (<=2-D tensor).
// ---------------------------------------------------------------------------
__device__ __forceinline__ void tdm_load_2d_bf16(
    unsigned lds_off, const void* gptr,
    unsigned tile_d0, unsigned tile_d1,
    unsigned tensor_d0, unsigned tensor_d1, unsigned stride0) {
  unsigned long long ga = (unsigned long long)gptr;
  u32x4 g0;
  g0[0] = 1u;                                           // count=1, user D#
  g0[1] = lds_off;                                      // lds_addr (bytes)
  g0[2] = (unsigned)ga;                                 // global_addr[31:0]
  g0[3] = (unsigned)((ga >> 32) & 0x01ffffffu) | (2u << 30);  // ga[56:32] | type=2
  u32x8 g1;
  g1[0] = 1u << 16;                                     // data_size=2B, wg_mask=0
  g1[1] = (tensor_d0 & 0xffffu) << 16;                  // tensor_dim0[15:0]
  g1[2] = (tensor_d0 >> 16) | ((tensor_d1 & 0xffffu) << 16);
  g1[3] = (tensor_d1 >> 16) | ((tile_d0 & 0xffffu) << 16);  // tile_dim0
  g1[4] = tile_d1 & 0xffffu;                            // tile_dim1 (tile_dim2=0)
  g1[5] = stride0;                                      // tensor_dim0_stride[31:0]
  g1[6] = 0u;
  g1[7] = 0u;
  asm volatile("tensor_load_to_lds %0, %1" :: "s"(g0), "s"(g1) : "memory");
}

__device__ __forceinline__ void tdm_wait0() {
#if __has_builtin(__builtin_amdgcn_s_wait_tensorcnt)
  __builtin_amdgcn_s_wait_tensorcnt(0);
#else
  asm volatile("s_wait_tensorcnt 0x0" ::: "memory");
#endif
}

__device__ __forceinline__ unsigned lds_offset_of(const void* p) {
  return (unsigned)(unsigned long long)p;   // LDS aperture: addr[31:0] == LDS offset
}

// ---------------------------------------------------------------------------
// Kernel 1: QKV GEMM  C[4096,2304] = x @ W_qkv^T + b, scatter to per-head bf16
// ---------------------------------------------------------------------------
__global__ __launch_bounds__(256) void qkv_kernel(
    const float* __restrict__ x, const float* __restrict__ Wqkv,
    const float* __restrict__ bqkv,
    bf16* __restrict__ qbuf, bf16* __restrict__ kbuf, bf16* __restrict__ vbuf) {
  __shared__ bf16 ldsX[64 * 32];
  __shared__ bf16 ldsW[64 * 32];
  int colBase = blockIdx.x * 64;
  int rowBase = blockIdx.y * 64;
  int t = threadIdx.x;
  int w = t >> 5, lane = t & 31;
  int tm = w & 3, tn0 = (w >> 2) * 2;
  int sr = t >> 2, sc = (t & 3) * 8;
  v8f acc[2] = {};

  for (int k0 = 0; k0 < DIMC; k0 += 32) {
    const float* xs = x    + (size_t)(rowBase + sr) * DIMC + k0 + sc;
    const float* ws = Wqkv + (size_t)(colBase + sr) * DIMC + k0 + sc;
    __builtin_prefetch(xs + 32, 0, 3);
    __builtin_prefetch(ws + 32, 0, 3);
    float4 x0 = ((const float4*)xs)[0];
    float4 x1 = ((const float4*)xs)[1];
    float4 w0 = ((const float4*)ws)[0];
    float4 w1 = ((const float4*)ws)[1];
    bf16* px = ldsX + sr * 32 + sc;
    bf16* pw = ldsW + sr * 32 + sc;
    px[0] = (bf16)x0.x; px[1] = (bf16)x0.y; px[2] = (bf16)x0.z; px[3] = (bf16)x0.w;
    px[4] = (bf16)x1.x; px[5] = (bf16)x1.y; px[6] = (bf16)x1.z; px[7] = (bf16)x1.w;
    pw[0] = (bf16)w0.x; pw[1] = (bf16)w0.y; pw[2] = (bf16)w0.z; pw[3] = (bf16)w0.w;
    pw[4] = (bf16)w1.x; pw[5] = (bf16)w1.y; pw[6] = (bf16)w1.z; pw[7] = (bf16)w1.w;
    __syncthreads();
    v16bf a  = load_a(ldsX + tm * 16 * 32, 32);
    acc[0] = wmma_bf16(a, load_bt(ldsW + (tn0 + 0) * 16 * 32, 32), acc[0]);
    acc[1] = wmma_bf16(a, load_bt(ldsW + (tn0 + 1) * 16 * 32, 32), acc[1]);
    __syncthreads();
  }

  int nl = lane & 15, hh = lane >> 4;
#pragma unroll
  for (int s = 0; s < 2; ++s) {
    int gcol = colBase + (tn0 + s) * 16 + nl;
    int sec  = gcol / DIMC;
    int dcol = gcol - sec * DIMC;
    int head = dcol >> 6, hd = dcol & 63;
    float bias = bqkv[gcol];
#pragma unroll
    for (int r = 0; r < 8; ++r) {
      int grow = rowBase + tm * 16 + r + 8 * hh;
      int bI = grow >> 10, nS = grow & 1023;
      float v = acc[s][r] + bias;
      size_t di = ((size_t)((bI * NH + head) << 10) + nS) * HD + hd;
      if (sec == 0)      qbuf[di] = (bf16)(v * QSCALE);
      else if (sec == 1) kbuf[di] = (bf16)v;
      else               vbuf[di] = (bf16)v;
    }
  }
}

// ---------------------------------------------------------------------------
// Kernel 2: gw = softmax(gelu_exact(v @ gp[h]^T)) over GP axis, store bf16
// ---------------------------------------------------------------------------
__global__ __launch_bounds__(256) void groupsim_kernel(
    const bf16* __restrict__ vbuf, const float* __restrict__ Wgp,
    bf16* __restrict__ gwbuf) {
  __shared__ bf16  ldsGP[64 * 64];
  __shared__ float ldsC[64 * 64];
  int bh = blockIdx.x;
  int h  = bh % NH;
  int rowBase = blockIdx.y * 64;
  int t = threadIdx.x;

  {  // stage gp[h] (64x64 f32 -> bf16)
    int e = t * 16;
    const float* src = Wgp + (size_t)h * 4096 + e;
#pragma unroll
    for (int i = 0; i < 16; ++i) ldsGP[e + i] = (bf16)src[i];
  }
  __syncthreads();

  int w = t >> 5;
  int tm = w & 3, tn0 = (w >> 2) * 2;
  const bf16* vrow = vbuf + ((size_t)(bh << 10) + rowBase + tm * 16) * HD;
  v16bf a0 = load_a(vrow, HD);
  v16bf a1 = load_a(vrow + 32, HD);
#pragma unroll
  for (int s = 0; s < 2; ++s) {
    int tn = tn0 + s;
    v8f acc = {};
    acc = wmma_bf16(a0, load_bt(ldsGP + tn * 16 * 64,      64), acc);
    acc = wmma_bf16(a1, load_bt(ldsGP + tn * 16 * 64 + 32, 64), acc);
    store_d_f32(ldsC + tm * 16 * 64 + tn * 16, 64, acc);
  }
  __syncthreads();

  if (t < 64) {
    float* row = ldsC + t * 64;
    float mx = -3.4e38f;
    for (int m = 0; m < GPC; ++m) {
      float xv = row[m];
      float g  = 0.5f * xv * (1.0f + erff(xv * 0.70710678118f));  // exact GELU
      row[m] = g;
      mx = fmaxf(mx, g);
    }
    float sum = 0.f;
    for (int m = 0; m < GPC; ++m) { float e = __expf(row[m] - mx); row[m] = e; sum += e; }
    float inv = 1.0f / sum;
    bf16* dst = gwbuf + ((size_t)(bh << 10) + rowBase + t) * GPC;
    for (int m = 0; m < GPC; ++m) dst[m] = (bf16)(row[m] * inv);
  }
}

// ---------------------------------------------------------------------------
// Kernel 3: Z = (q@k^T)*(gw@gw^T), row softmax, A = (1-a)*P + a*G -> Abuf
// ---------------------------------------------------------------------------
__global__ __launch_bounds__(256) void attngate_kernel(
    const bf16* __restrict__ qbuf, const bf16* __restrict__ kbuf,
    const bf16* __restrict__ gwbuf, const float* __restrict__ alpha,
    float* __restrict__ Abuf) {
  __shared__ float pbuf[2048];
  __shared__ float rowmax[16];
  __shared__ float rowsum[16];
  int bh = blockIdx.x;
  int h  = bh % NH;
  int rowBase = blockIdx.y * 16;
  int t = threadIdx.x;
  int w = t >> 5, lane = t & 31;
  int hh = lane >> 4, nl = lane & 15;

  const bf16* qrow  = qbuf  + ((size_t)(bh << 10) + rowBase) * HD;
  const bf16* gwrow = gwbuf + ((size_t)(bh << 10) + rowBase) * GPC;
  v16bf aq0 = load_a(qrow, HD),  aq1 = load_a(qrow + 32, HD);
  v16bf ag0 = load_a(gwrow, GPC), ag1 = load_a(gwrow + 32, GPC);

  v8f zs[8], gs[8];
#pragma unroll
  for (int j = 0; j < 8; ++j) {
    int c0 = (w * 8 + j) * 16;
    const bf16* kc = kbuf  + ((size_t)(bh << 10) + c0) * HD;
    const bf16* gc = gwbuf + ((size_t)(bh << 10) + c0) * GPC;
    v8f s = {};
    s = wmma_bf16(aq0, load_bt(kc,      HD), s);
    s = wmma_bf16(aq1, load_bt(kc + 32, HD), s);
    v8f g = {};
    g = wmma_bf16(ag0, load_bt(gc,      GPC), g);
    g = wmma_bf16(ag1, load_bt(gc + 32, GPC), g);
    gs[j] = g;
    v8f z;
#pragma unroll
    for (int r = 0; r < 8; ++r) z[r] = s[r] * g[r];
    zs[j] = z;
  }

#pragma unroll
  for (int r = 0; r < 8; ++r) {
    float pm = -3.4e38f;
#pragma unroll
    for (int j = 0; j < 8; ++j) pm = fmaxf(pm, zs[j][r]);
    pbuf[w * 256 + lane * 8 + r] = pm;
  }
  __syncthreads();
  if (t < 16) {
    int half = t >> 3, r = t & 7;
    float mx = -3.4e38f;
    for (int ww = 0; ww < 8; ++ww)
      for (int l = 0; l < 16; ++l)
        mx = fmaxf(mx, pbuf[ww * 256 + (half * 16 + l) * 8 + r]);
    rowmax[t] = mx;
  }
  __syncthreads();

#pragma unroll
  for (int r = 0; r < 8; ++r) {
    int row = r + 8 * hh;
    float ps = 0.f;
#pragma unroll
    for (int j = 0; j < 8; ++j) ps += __expf(zs[j][r] - rowmax[row]);
    pbuf[w * 256 + lane * 8 + r] = ps;
  }
  __syncthreads();
  if (t < 16) {
    int half = t >> 3, r = t & 7;
    float sm = 0.f;
    for (int ww = 0; ww < 8; ++ww)
      for (int l = 0; l < 16; ++l)
        sm += pbuf[ww * 256 + (half * 16 + l) * 8 + r];
    rowsum[t] = sm;
  }
  __syncthreads();

  float aH = 1.0f / (1.0f + __expf(-alpha[h]));
  float* arow = Abuf + (size_t)bh * 1048576;
#pragma unroll
  for (int j = 0; j < 8; ++j) {
    int c0 = (w * 8 + j) * 16;
#pragma unroll
    for (int r = 0; r < 8; ++r) {
      int row = r + 8 * hh;
      float P  = __expf(zs[j][r] - rowmax[row]) / rowsum[row];
      float Av = (1.0f - aH) * P + aH * gs[j][r];
      arow[(size_t)(rowBase + row) * 1024 + c0 + nl] = Av;
    }
  }
}

// ---------------------------------------------------------------------------
// Kernel 4: colsum[bh][m] = sum_n A[bh][n][m]
// ---------------------------------------------------------------------------
__global__ __launch_bounds__(256) void colsum_kernel(
    const float* __restrict__ Abuf, float* __restrict__ colsum) {
  int g  = blockIdx.x * 256 + threadIdx.x;
  int bh = g >> 10;
  int m  = g & 1023;
  const float* base = Abuf + (size_t)bh * 1048576 + m;
  float s = 0.f;
  for (int n = 0; n < SEQN; ++n) {
    __builtin_prefetch(base + (size_t)(n + 16) * 1024, 0, 1);
    s += base[(size_t)n * 1024];
  }
  colsum[g] = s;
}

// ---------------------------------------------------------------------------
// Kernel 5: O = (A / (colsum+1e-8)) @ v ; v tile staged by the TDM (4KB/step)
// ---------------------------------------------------------------------------
__global__ __launch_bounds__(256) void av_kernel(
    const float* __restrict__ Abuf, const float* __restrict__ colsum,
    const bf16* __restrict__ vbuf, bf16* __restrict__ attn_out) {
  __shared__ bf16 ldsA[64 * 32];
  __shared__ bf16 ldsV[32 * 64];
  int bh = blockIdx.x;
  int b  = bh / NH, head = bh % NH;
  int rowBase = blockIdx.y * 64;
  int t = threadIdx.x;
  int w = t >> 5, lane = t & 31;
  int tm = w & 3, tn0 = (w >> 2) * 2;
  int sr = t >> 2, sc = (t & 3) * 8;
  v8f acc[2] = {};
  const float* csb = colsum + (bh << 10);
  unsigned ldsV_off = lds_offset_of(ldsV);

  for (int k0 = 0; k0 < SEQN; k0 += 32) {
    if (w == 0) {  // async DMA: 32x64 bf16 tile is one contiguous 4KB block
      tdm_load_2d_bf16(ldsV_off, vbuf + ((size_t)(bh << 10) + k0) * HD,
                       /*tile_d0=*/2048, /*tile_d1=*/1,
                       /*tensor_d0=*/2048, /*tensor_d1=*/1, /*stride0=*/2048);
    }
    {  // stage A / colsum -> bf16 (needs VALU conversion, stays manual)
      const float* as = Abuf + (size_t)bh * 1048576 + (size_t)(rowBase + sr) * 1024 + k0 + sc;
      const float* cs = csb + k0 + sc;
      float4 a0 = ((const float4*)as)[0];
      float4 a1 = ((const float4*)as)[1];
      float4 c0 = ((const float4*)cs)[0];
      float4 c1 = ((const float4*)cs)[1];
      bf16* pa = ldsA + sr * 32 + sc;
      pa[0] = (bf16)(a0.x / (c0.x + 1e-8f)); pa[1] = (bf16)(a0.y / (c0.y + 1e-8f));
      pa[2] = (bf16)(a0.z / (c0.z + 1e-8f)); pa[3] = (bf16)(a0.w / (c0.w + 1e-8f));
      pa[4] = (bf16)(a1.x / (c1.x + 1e-8f)); pa[5] = (bf16)(a1.y / (c1.y + 1e-8f));
      pa[6] = (bf16)(a1.z / (c1.z + 1e-8f)); pa[7] = (bf16)(a1.w / (c1.w + 1e-8f));
    }
    tdm_wait0();
    __syncthreads();
    v16bf a = load_a(ldsA + tm * 16 * 32, 32);
    acc[0] = wmma_bf16(a, load_b(ldsV + (tn0 + 0) * 16, 64), acc[0]);
    acc[1] = wmma_bf16(a, load_b(ldsV + (tn0 + 1) * 16, 64), acc[1]);
    __syncthreads();
  }

  int nl = lane & 15, hh = lane >> 4;
#pragma unroll
  for (int s = 0; s < 2; ++s) {
    int col = (tn0 + s) * 16 + nl;
#pragma unroll
    for (int r = 0; r < 8; ++r) {
      int row = rowBase + tm * 16 + r + 8 * hh;
      attn_out[((size_t)(b << 10) + row) * DIMC + (head << 6) + col] = (bf16)acc[s][r];
    }
  }
}

// ---------------------------------------------------------------------------
// Kernel 6: out = attn_out @ W_proj^T + b_proj ; A tile staged by the TDM
// (64 rows x 32 bf16 cols, row stride 768 elements -> 2-D D# descriptor)
// ---------------------------------------------------------------------------
__global__ __launch_bounds__(256) void proj_kernel(
    const bf16* __restrict__ attn_out, const float* __restrict__ Wp,
    const float* __restrict__ bp, float* __restrict__ out) {
  __shared__ bf16 ldsA[64 * 32];
  __shared__ bf16 ldsW[64 * 32];
  int colBase = blockIdx.x * 64;
  int rowBase = blockIdx.y * 64;
  int t = threadIdx.x;
  int w = t >> 5, lane = t & 31;
  int tm = w & 3, tn0 = (w >> 2) * 2;
  int sr = t >> 2, sc = (t & 3) * 8;
  v8f acc[2] = {};
  unsigned ldsA_off = lds_offset_of(ldsA);

  for (int k0 = 0; k0 < DIMC; k0 += 32) {
    if (w == 0) {  // async DMA of the strided activation tile
      tdm_load_2d_bf16(ldsA_off, attn_out + (size_t)rowBase * DIMC + k0,
                       /*tile_d0=*/32, /*tile_d1=*/64,
                       /*tensor_d0=*/DIMC, /*tensor_d1=*/BDIM * SEQN,
                       /*stride0=*/DIMC);
    }
    const float* wsrc = Wp + (size_t)(colBase + sr) * DIMC + k0 + sc;
    __builtin_prefetch(wsrc + 32, 0, 3);
    float4 w0 = ((const float4*)wsrc)[0];
    float4 w1 = ((const float4*)wsrc)[1];
    bf16* pw = ldsW + sr * 32 + sc;
    pw[0] = (bf16)w0.x; pw[1] = (bf16)w0.y; pw[2] = (bf16)w0.z; pw[3] = (bf16)w0.w;
    pw[4] = (bf16)w1.x; pw[5] = (bf16)w1.y; pw[6] = (bf16)w1.z; pw[7] = (bf16)w1.w;
    tdm_wait0();
    __syncthreads();
    v16bf a = load_a(ldsA + tm * 16 * 32, 32);
    acc[0] = wmma_bf16(a, load_bt(ldsW + (tn0 + 0) * 16 * 32, 32), acc[0]);
    acc[1] = wmma_bf16(a, load_bt(ldsW + (tn0 + 1) * 16 * 32, 32), acc[1]);
    __syncthreads();
  }

  int nl = lane & 15, hh = lane >> 4;
#pragma unroll
  for (int s = 0; s < 2; ++s) {
    int gcol = colBase + (tn0 + s) * 16 + nl;
    float bias = bp[gcol];
#pragma unroll
    for (int r = 0; r < 8; ++r) {
      int grow = rowBase + tm * 16 + r + 8 * hh;
      out[(size_t)grow * DIMC + gcol] = acc[s][r] + bias;
    }
  }
}

// ---------------------------------------------------------------------------
extern "C" void kernel_launch(void* const* d_in, const int* in_sizes, int n_in,
                              void* d_out, int out_size, void* d_ws, size_t ws_size,
                              hipStream_t stream) {
  (void)in_sizes; (void)n_in; (void)out_size; (void)ws_size;
  const float* x     = (const float*)d_in[0];
  const float* Wqkv  = (const float*)d_in[1];
  const float* bqkv  = (const float*)d_in[2];
  const float* Wgp   = (const float*)d_in[3];
  const float* alpha = (const float*)d_in[4];
  const float* Wproj = (const float*)d_in[5];
  const float* bproj = (const float*)d_in[6];
  float* out = (float*)d_out;

  char* ws = (char*)d_ws;
  const size_t headElems = (size_t)BDIM * NH * SEQN * HD;   // 3,145,728
  bf16* qbuf     = (bf16*)ws;  ws += headElems * 2;
  bf16* kbuf     = (bf16*)ws;  ws += headElems * 2;
  bf16* vbuf     = (bf16*)ws;  ws += headElems * 2;
  bf16* gwbuf    = (bf16*)ws;  ws += headElems * 2;
  bf16* attn_out = (bf16*)ws;  ws += (size_t)BDIM * SEQN * DIMC * 2;
  float* colsum  = (float*)ws; ws += (size_t)BDIM * NH * SEQN * 4;
  float* Abuf    = (float*)ws;  // 201 MB: [B*H][N][N] f32

  dim3 blk(256);
  qkv_kernel     <<<dim3(36, 64), blk, 0, stream>>>(x, Wqkv, bqkv, qbuf, kbuf, vbuf);
  groupsim_kernel<<<dim3(48, 16), blk, 0, stream>>>(vbuf, Wgp, gwbuf);
  attngate_kernel<<<dim3(48, 64), blk, 0, stream>>>(qbuf, kbuf, gwbuf, alpha, Abuf);
  colsum_kernel  <<<dim3(192),    blk, 0, stream>>>(Abuf, colsum);
  av_kernel      <<<dim3(48, 16), blk, 0, stream>>>(Abuf, colsum, vbuf, attn_out);
  proj_kernel    <<<dim3(12, 64), blk, 0, stream>>>(attn_out, Wproj, bproj, out);
}
